// CRFasRNN_65627100283468
// MI455X (gfx1250) — matrix operand order repaired and checked by reading, not compiled, not run on hardware
//
#include <hip/hip_runtime.h>

// ---------------------------------------------------------------------------
// CRF-as-RNN mean-field on MI455X (gfx1250, wave32).
// 19-tap separable Gaussian blur on the WMMA pipe:
//   16x16 output tile = two chained v_wmma_f32_16x16x32_f16 (two K=32 slabs
//   cover the 34-element footprint; out-of-band weights are zero).
// Gaussian band matrices are compile-time __constant__ per-lane fragment
// tables (2x16B loads/lane). The h-blur writes its intermediate column-major
// so the v-blur's B operand is two aligned uint4 loads per lane (no gather).
// ---------------------------------------------------------------------------

typedef __attribute__((ext_vector_type(16))) _Float16 v16h;
typedef __attribute__((ext_vector_type(8)))  float    v8f;
typedef _Float16 h16;

static constexpr int IMG_H = 2048;
static constexpr int IMG_W = 2048;
static constexpr int TILES = 128;            // 2048 / 16
static constexpr size_t NPIX = (size_t)IMG_H * IMG_W;

// Normalized Gaussian, sigma = 3, radius = 9 (matches _gauss_kernel_1d).
constexpr float kGW[10] = {
    0.13317620f, 0.12597893f, 0.10663903f, 0.08077548f, 0.05475031f,
    0.03320770f, 0.01802341f, 0.00875346f, 0.00380397f, 0.00147945f};

constexpr float gwc(int t) {
  int a = t < 0 ? -t : t;
  return (a <= 9) ? kGW[a] : 0.0f;
}

// Per-lane WMMA fragment tables, built at compile time.
struct alignas(32) FragTab { _Float16 v[32][16]; };

// B-operand layout (32x16): lane -> col n = lane&15, K = (j<8 ? j : j+8) + (lane>>4)*8.
constexpr FragTab make_b(int shift) {
  FragTab t{};
  for (int lane = 0; lane < 32; ++lane) {
    int n = lane & 15, kb = (lane >> 4) * 8;
    for (int j = 0; j < 16; ++j)
      t.v[lane][j] = (_Float16)gwc((j < 8 ? j : j + 8) + kb + shift - n);
  }
  return t;
}

// A-operand layout (16x32): lane -> row m = lane&15,
// K = (lane>>4)*16 + j.
constexpr FragTab make_a(int shift) {
  FragTab t{};
  for (int lane = 0; lane < 32; ++lane) {
    int m = lane & 15, kb = (lane >> 4) * 16;
    for (int j = 0; j < 16; ++j) {
      int k = j + kb;
      t.v[lane][j] = (_Float16)gwc(k + shift - m);
    }
  }
  return t;
}

__device__ __constant__ FragTab HB1 = make_b(-16);  // h-blur B, cols c0-16..c0+15
__device__ __constant__ FragTab HB2 = make_b(+16);  // h-blur B, cols c0+16..c0+47
__device__ __constant__ FragTab VA1 = make_a(-16);  // v-blur A, rows r0-16..r0+15
__device__ __constant__ FragTab VA2 = make_a(+16);  // v-blur A, rows r0+16..r0+47

__device__ __forceinline__ v16h load_frag(const FragTab& t, int lane) {
  return *(const v16h*)t.v[lane];                   // 32B, 32B-aligned
}

// ---------------------------------------------------------------------------
// Kernel 1: unaries + initial softmax.  U0/U1 f32, S f16 (2 ch, row-major).
// ---------------------------------------------------------------------------
__global__ void k_unary(const float* __restrict__ img,
                        const int* __restrict__ anno,
                        float* __restrict__ U0, float* __restrict__ U1,
                        h16* __restrict__ S) {
  size_t i = (size_t)blockIdx.x * blockDim.x + threadIdx.x;
  float v = img[i];                     // image[0] = first H*W floats of (3,H,W)
  int a = anno[i];
  float d0 = (v - 10.0f)  * (v - 10.0f)  * 1e-4f;
  float d1 = (v - 120.0f) * (v - 120.0f) * 1e-4f;
  const float PE = 0.69314718f;         // -log(0.5)
  const float NE = 1.38629436f;         // -log(0.25)
  float u0, u1;
  if (a == 0)      { u0 = PE + d0; u1 = NE + d1; }
  else if (a == 1) { u0 = NE + d0; u1 = PE + d1; }
  else             { u0 = 0.0f;    u1 = 0.0f;    }
  U0[i] = u0;
  U1[i] = u1;
  float m  = fmaxf(u0, u1);
  float e0 = __expf(u0 - m), e1 = __expf(u1 - m);
  float s0 = e0 / (e0 + e1);
  S[i]        = (h16)s0;
  S[NPIX + i] = (h16)(1.0f - s0);
}

// ---------------------------------------------------------------------------
// Kernel 2: horizontal blur via WMMA.  One wave = one 16x16 tile of one
// channel.  A = data rows (row-major S), B = constant band matrix.
// Output written COLUMN-MAJOR into T (one uint4 store per lane).
// ---------------------------------------------------------------------------
__global__ void k_hblur(const h16* __restrict__ S, h16* __restrict__ T) {
  int lane = threadIdx.x & 31;
  int wid  = threadIdx.x >> 5;
  int gw_id = blockIdx.x * 8 + wid;               // 128*128*2 waves total
  int tx = gw_id & (TILES - 1);
  int ty = (gw_id >> 7) & (TILES - 1);
  int ch = gw_id >> 14;
  int r0 = ty * 16, c0 = tx * 16;
  const h16* src = S + (size_t)ch * NPIX;

  // A fragments (data): lane row = lane&15; K chunks kb..kb+7 and kb+16..kb+23.
  int row = lane & 15;
  int kb  = (lane >> 4) * 8;
  const h16* base = src + (size_t)(r0 + row) * IMG_W + (c0 - 16);
  v16h a1, a2;
  if (c0 >= 16 && c0 + 48 <= IMG_W) {              // wave-uniform interior path
    union { v16h h; uint4 u[2]; } ua, ub;
    ua.u[0] = *(const uint4*)(base + kb);
    ua.u[1] = *(const uint4*)(base + kb + 16);
    ub.u[0] = *(const uint4*)(base + kb + 32);
    ub.u[1] = *(const uint4*)(base + kb + 48);
    a1 = ua.h; a2 = ub.h;
  } else {                                         // border: zero-pad ('SAME')
#pragma unroll
    for (int j = 0; j < 16; ++j) {
      int k  = (j < 8 ? j : j + 8) + kb;
      int c1 = c0 - 16 + k;
      int c2 = c1 + 32;
      a1[j] = (c1 >= 0 && c1 < IMG_W)
                  ? src[(size_t)(r0 + row) * IMG_W + c1] : (h16)0.0f;
      a2[j] = (c2 >= 0 && c2 < IMG_W)
                  ? src[(size_t)(r0 + row) * IMG_W + c2] : (h16)0.0f;
    }
  }

  v16h b1 = load_frag(HB1, lane);
  v16h b2 = load_frag(HB2, lane);

  v8f acc = {};
  acc = __builtin_amdgcn_wmma_f32_16x16x32_f16(false, a1, false, b1,
                                               (short)0, acc, false, false);
  acc = __builtin_amdgcn_wmma_f32_16x16x32_f16(false, a2, false, b2,
                                               (short)0, acc, false, false);

  // D: VGPR i -> row rbase+i, col c0+n.  Column-major T => contiguous 8 halves.
  int n = lane & 15;
  int rbase = r0 + (lane >> 4) * 8;
  h16* dstcol = T + (size_t)ch * NPIX + (size_t)(c0 + n) * IMG_H + rbase;
  union { uint4 u; h16 h[8]; } pk;
#pragma unroll
  for (int i = 0; i < 8; ++i) pk.h[i] = (h16)acc[i];
  *(uint4*)dstcol = pk.u;                          // 16B-aligned (rbase % 8 == 0)
}

// ---------------------------------------------------------------------------
// Kernel 3: vertical blur via WMMA (A = constant band, B = column-major data)
// for both channels + fused 2x2 affine + q + softmax (+ final threshold).
// One wave = one 16x16 pixel tile, both channels.
// ---------------------------------------------------------------------------
__global__ void k_vfused(const h16* __restrict__ T,
                         const float* __restrict__ U0, const float* __restrict__ U1,
                         const float* __restrict__ sw, const float* __restrict__ sb,
                         const float* __restrict__ cw, const float* __restrict__ cb,
                         h16* __restrict__ S, float* __restrict__ OUT, int last) {
  int lane = threadIdx.x & 31;
  int wid  = threadIdx.x >> 5;
  int gid  = blockIdx.x * 8 + wid;                 // 128*128 waves
  int tx = gid & (TILES - 1);
  int ty = gid >> 7;
  int r0 = ty * 16, c0 = tx * 16;

  v16h A1 = load_frag(VA1, lane);
  v16h A2 = load_frag(VA2, lane);

  int n    = lane & 15;
  int kb16 = (lane >> 4) * 16;
  int col  = c0 + n;
  int rA   = r0 - 16 + kb16;                       // first row of this lane's B1

  v8f acc[2];
#pragma unroll
  for (int ch = 0; ch < 2; ++ch) {
    const h16* colbase = T + (size_t)ch * NPIX + (size_t)col * IMG_H;
    v16h B1, B2;
    if (r0 >= 16 && r0 + 48 <= IMG_H) {            // wave-uniform interior path
      union { v16h h; uint4 u[2]; } u1, u2;
      u1.u[0] = *(const uint4*)(colbase + rA);
      u1.u[1] = *(const uint4*)(colbase + rA + 8);
      u2.u[0] = *(const uint4*)(colbase + rA + 32);
      u2.u[1] = *(const uint4*)(colbase + rA + 40);
      B1 = u1.h; B2 = u2.h;
    } else {                                       // border: zero-pad ('SAME')
#pragma unroll
      for (int j = 0; j < 16; ++j) {
        int ra = rA + j;
        int rb = ra + 32;
        B1[j] = (ra >= 0 && ra < IMG_H) ? colbase[ra] : (h16)0.0f;
        B2[j] = (rb >= 0 && rb < IMG_H) ? colbase[rb] : (h16)0.0f;
      }
    }
    v8f c = {};
    c = __builtin_amdgcn_wmma_f32_16x16x32_f16(false, A1, false, B1,
                                               (short)0, c, false, false);
    c = __builtin_amdgcn_wmma_f32_16x16x32_f16(false, A2, false, B2,
                                               (short)0, c, false, false);
    acc[ch] = c;
  }

  // Fuse the two 1x1 convs: pairwise = (cw@sw)*blur + (cw@sb + cb)
  float f00 = cw[0] * sw[0] + cw[1] * sw[2];
  float f01 = cw[0] * sw[1] + cw[1] * sw[3];
  float f10 = cw[2] * sw[0] + cw[3] * sw[2];
  float f11 = cw[2] * sw[1] + cw[3] * sw[3];
  float fb0 = cw[0] * sb[0] + cw[1] * sb[1] + cb[0];
  float fb1 = cw[2] * sb[0] + cw[3] * sb[1] + cb[1];

  int rbase = r0 + (lane >> 4) * 8;
#pragma unroll
  for (int i = 0; i < 8; ++i) {
    size_t pix = (size_t)(rbase + i) * IMG_W + col;
    float b0 = acc[0][i], b1v = acc[1][i];
    float p0 = f00 * b0 + f01 * b1v + fb0;
    float p1 = f10 * b0 + f11 * b1v + fb1;
    float q0 = U0[pix] - p0;
    float q1 = U1[pix] - p1;
    float mx = fmaxf(q0, q1);
    float e0 = __expf(q0 - mx), e1 = __expf(q1 - mx);
    float s0 = e0 / (e0 + e1);
    if (last) {
      // t = s>0.5 ? s : 0 ; y = (-t > -1e-6) ? -t : 1  ==> binary mask
      OUT[pix]        = (s0 > 0.5f) ? 1.0f : 0.0f;
      OUT[NPIX + pix] = ((1.0f - s0) > 0.5f) ? 1.0f : 0.0f;
    } else {
      S[pix]        = (h16)s0;
      S[NPIX + pix] = (h16)(1.0f - s0);
    }
  }
}

// ---------------------------------------------------------------------------
extern "C" void kernel_launch(void* const* d_in, const int* in_sizes, int n_in,
                              void* d_out, int out_size, void* d_ws, size_t ws_size,
                              hipStream_t stream) {
  const float* img  = (const float*)d_in[0];   // (3,H,W); only channel 0 used
  // d_in[1] = features: dead code in reference (bilateral branch unused)
  const int*   anno = (const int*)d_in[2];     // (H,W) int32
  const float* sw   = (const float*)d_in[3];   // (2,2)
  const float* sb   = (const float*)d_in[4];   // (2,)
  const float* cw   = (const float*)d_in[5];   // (2,2)
  const float* cb   = (const float*)d_in[6];   // (2,)
  float* out = (float*)d_out;                  // (1,2,H,W) f32

  // Workspace: U0 | U1 (f32, 16MB each) | S (f16 2ch, 16MB) | T (f16 2ch, 16MB)
  char* ws = (char*)d_ws;
  float* U0 = (float*)(ws);
  float* U1 = (float*)(ws + (size_t)16 * 1024 * 1024);
  h16*   S  = (h16*)  (ws + (size_t)32 * 1024 * 1024);
  h16*   T  = (h16*)  (ws + (size_t)48 * 1024 * 1024);  // column-major per channel

  k_unary<<<(int)(NPIX / 256), 256, 0, stream>>>(img, anno, U0, U1, S);

  for (int it = 0; it < 5; ++it) {
    // 128*128 tiles * 2 channels, 8 waves per 256-thread block
    k_hblur<<<(TILES * TILES * 2) / 8, 256, 0, stream>>>(S, T);
    // 128*128 tiles (both channels per wave)
    k_vfused<<<(TILES * TILES) / 8, 256, 0, stream>>>(T, U0, U1, sw, sb, cw, cb,
                                                      S, out, it == 4 ? 1 : 0);
  }
}
